// decoderRNN_47313359733311
// MI455X (gfx1250) — compile-verified, matrix operand order checked by reading, not run on hardware
//
#include <hip/hip_runtime.h>
#include <hip/hip_bf16.h>

#define BB 64
#define TKK 2048
#define HH 512
#define EE 128
#define VV 50000
#define NOOV 512
#define H2 1024
#define H4 2048
#define VEXT (VV + NOOV)

typedef __attribute__((ext_vector_type(16))) __bf16 v16bf;
typedef __attribute__((ext_vector_type(8)))  float  v8f;

static __device__ __forceinline__ float sigmf(float x) { return 1.0f / (1.0f + expf(-x)); }

static __device__ __forceinline__ __bf16 f2bf(float f) {
    union { float f; unsigned u; } v; v.f = f;
    unsigned r = v.u + 0x7FFFu + ((v.u >> 16) & 1u);  // RNE
    unsigned short h = (unsigned short)(r >> 16);
    return __builtin_bit_cast(__bf16, h);
}

// ---------------- K1: embedding gather + context projection -> x[B,E] ---------
__global__ void k_ctx_proj(const int* __restrict__ ids, const float* __restrict__ W_emb,
                           const float* __restrict__ ct1, const float* __restrict__ W_ctx,
                           const float* __restrict__ b_ctx, float* __restrict__ xout) {
    __shared__ float s_in[H2 + EE];
    const int b = blockIdx.x;
    const int id = ids[b];
    for (int i = threadIdx.x; i < H2; i += 128) s_in[i] = ct1[b * H2 + i];
    for (int i = threadIdx.x; i < EE; i += 128) s_in[H2 + i] = W_emb[(size_t)id * EE + i];
    __syncthreads();
    const int e = threadIdx.x;
    float acc = b_ctx[e];
    const float* w = W_ctx + (size_t)e * (H2 + EE);
    for (int k = 0; k < H2 + EE; ++k) acc += s_in[k] * w[k];
    xout[b * EE + e] = acc;
}

// ---------------- K2: fused LSTM cell ----------------------------------------
__global__ void k_lstm(const float* __restrict__ x, const float* __restrict__ h0,
                       const float* __restrict__ c0, const float* __restrict__ W_ih,
                       const float* __restrict__ W_hh, const float* __restrict__ b_ih,
                       const float* __restrict__ b_hh, float* __restrict__ hdec,
                       float* __restrict__ cdec) {
    __shared__ float s_x[EE];
    __shared__ float s_h[HH];
    const int g = blockIdx.x * 256 + threadIdx.x;  // 32768 threads
    const int b = g >> 9;                          // constant within block
    const int h = g & 511;
    for (int i = threadIdx.x; i < EE; i += 256) s_x[i] = x[b * EE + i];
    for (int i = threadIdx.x; i < HH; i += 256) s_h[i] = h0[b * HH + i];
    __syncthreads();
    float pre[4];
#pragma unroll
    for (int q = 0; q < 4; ++q) {
        const int j = q * HH + h;
        float acc = b_ih[j] + b_hh[j];
        const float* wi = W_ih + (size_t)j * EE;
        for (int k = 0; k < EE; ++k) acc += s_x[k] * wi[k];
        const float* wh = W_hh + (size_t)j * HH;
        for (int k = 0; k < HH; ++k) acc += s_h[k] * wh[k];
        pre[q] = acc;
    }
    const float ig = sigmf(pre[0]), fg = sigmf(pre[1]);
    const float gg = tanhf(pre[2]), og = sigmf(pre[3]);
    const float c = fg * c0[b * HH + h] + ig * gg;
    cdec[b * HH + h] = c;
    hdec[b * HH + h] = og * tanhf(c);
}

// ---------------- K3: dec_feature = h_c_dec @ W_attn^T + b ------------------
__global__ void k_decfeat(const float* __restrict__ hdec, const float* __restrict__ cdec,
                          const float* __restrict__ W_attn, const float* __restrict__ b_attn,
                          float* __restrict__ df) {
    const int g = blockIdx.x * 256 + threadIdx.x;  // 65536
    const int b = g >> 10;
    const int n = g & 1023;
    float acc = b_attn[n];
    const float* w = W_attn + (size_t)n * H2;
    for (int k = 0; k < HH; ++k) acc += hdec[b * HH + k] * w[k];
    for (int k = 0; k < HH; ++k) acc += cdec[b * HH + k] * w[HH + k];
    df[g] = acc;
}

// ---------------- K4: attention scores (HBM-streaming, 537 MB) ---------------
#define TT 64
__global__ void k_scores(const float* __restrict__ EF, const float* __restrict__ df,
                         const float* __restrict__ vat, float* __restrict__ scores) {
    __shared__ float s_df[H2];
    __shared__ float s_v[H2];
    const int b = blockIdx.y;
    const int tbase = blockIdx.x * TT;
    for (int i = threadIdx.x; i < H2; i += 256) {
        s_df[i] = df[b * H2 + i];
        s_v[i]  = vat[i];
    }
    __syncthreads();
    const int lane = threadIdx.x & 31;
    const int wave = threadIdx.x >> 5;
    for (int tt = wave; tt < TT; tt += 8) {
        const int t = tbase + tt;
        const float4* ef = (const float4*)(EF + (size_t)(b * TKK + t) * H2);
        __builtin_prefetch(EF + (size_t)(b * TKK + t + 8) * H2 + lane * 16, 0, 1);
        float acc = 0.f;
#pragma unroll
        for (int n0 = 0; n0 < H2 / 4; n0 += 32) {
            const float4 e = ef[n0 + lane];
            const int base = 4 * (n0 + lane);
            acc += tanhf(e.x + s_df[base + 0]) * s_v[base + 0];
            acc += tanhf(e.y + s_df[base + 1]) * s_v[base + 1];
            acc += tanhf(e.z + s_df[base + 2]) * s_v[base + 2];
            acc += tanhf(e.w + s_df[base + 3]) * s_v[base + 3];
        }
#pragma unroll
        for (int off = 16; off > 0; off >>= 1) acc += __shfl_xor(acc, off, 32);
        if (lane == 0) scores[b * TKK + t] = acc;
    }
}

// ---------------- K5: softmax over TK (+mask renorm), zero c_t, copy coverage -
__global__ void k_softmax_attn(const float* __restrict__ scores, const float* __restrict__ mask,
                               const float* __restrict__ cov_in, float* __restrict__ attn_out,
                               float* __restrict__ ct_out, float* __restrict__ cov_out) {
    __shared__ float red[256];
    const int b = blockIdx.x;
    const int tid = threadIdx.x;
    float m = -3.4e38f;
    for (int t = tid; t < TKK; t += 256) m = fmaxf(m, scores[b * TKK + t]);
    red[tid] = m; __syncthreads();
    for (int s = 128; s > 0; s >>= 1) { if (tid < s) red[tid] = fmaxf(red[tid], red[tid + s]); __syncthreads(); }
    m = red[0]; __syncthreads();
    float s = 0.f;
    for (int t = tid; t < TKK; t += 256) s += expf(scores[b * TKK + t] - m);
    red[tid] = s; __syncthreads();
    for (int q = 128; q > 0; q >>= 1) { if (tid < q) red[tid] += red[tid + q]; __syncthreads(); }
    const float S = red[0]; __syncthreads();
    float s2 = 0.f;
    for (int t = tid; t < TKK; t += 256)
        s2 += (expf(scores[b * TKK + t] - m) / S) * mask[b * TKK + t];
    red[tid] = s2; __syncthreads();
    for (int q = 128; q > 0; q >>= 1) { if (tid < q) red[tid] += red[tid + q]; __syncthreads(); }
    const float S2 = red[0]; __syncthreads();
    for (int t = tid; t < TKK; t += 256)
        attn_out[b * TKK + t] = (expf(scores[b * TKK + t] - m) / S) * mask[b * TKK + t] / S2;
    for (int i = tid; i < H2; i += 256) ct_out[b * H2 + i] = 0.f;
    for (int t = tid; t < TKK; t += 256) cov_out[b * TKK + t] = cov_in[b * TKK + t];
}

// ---------------- K6: c_t = attn @ encoder_outputs (HBM-streaming, 537 MB) ---
#define TSPLIT 8
__global__ void k_ctxvec(const float* __restrict__ EO, const float* __restrict__ attn,
                         float* __restrict__ ct_out) {
    __shared__ float s_a[TKK / TSPLIT];
    const int b = blockIdx.y;
    const int t0 = blockIdx.x * (TKK / TSPLIT);
    for (int i = threadIdx.x; i < TKK / TSPLIT; i += 256) s_a[i] = attn[b * TKK + t0 + i];
    __syncthreads();
    const int n = threadIdx.x * 4;
    float4 acc = {0.f, 0.f, 0.f, 0.f};
    for (int tt = 0; tt < TKK / TSPLIT; ++tt) {
        const float a = s_a[tt];
        __builtin_prefetch(EO + (size_t)(b * TKK + t0 + tt + 4) * H2 + n, 0, 1);
        const float4 e = *(const float4*)(EO + (size_t)(b * TKK + t0 + tt) * H2 + n);
        acc.x += a * e.x; acc.y += a * e.y; acc.z += a * e.z; acc.w += a * e.w;
    }
    atomicAdd(&ct_out[b * H2 + n + 0], acc.x);
    atomicAdd(&ct_out[b * H2 + n + 1], acc.y);
    atomicAdd(&ct_out[b * H2 + n + 2], acc.z);
    atomicAdd(&ct_out[b * H2 + n + 3], acc.w);
}

// ---------------- K7: p_gen + out1 -------------------------------------------
__global__ void k_pgen_out1(const float* __restrict__ hdec, const float* __restrict__ cdec,
                            const float* __restrict__ ct, const float* __restrict__ xbuf,
                            const float* __restrict__ W_out1, const float* __restrict__ b_out1,
                            const float* __restrict__ W_pgen, const float* __restrict__ b_pgen,
                            float* __restrict__ out1f, float* __restrict__ pgen_out) {
    __shared__ float s_q[3 * HH];
    __shared__ float red[256];
    const int b = blockIdx.x;
    const int tid = threadIdx.x;
    for (int i = tid; i < HH; i += 256) s_q[i] = hdec[b * HH + i];
    for (int i = tid; i < H2; i += 256) s_q[HH + i] = ct[b * H2 + i];
    __syncthreads();
    for (int h = tid; h < HH; h += 256) {
        float acc = b_out1[h];
        const float* w = W_out1 + (size_t)h * (3 * HH);
        for (int k = 0; k < 3 * HH; ++k) acc += s_q[k] * w[k];
        out1f[b * HH + h] = acc;
    }
    float p = 0.f;
    for (int k = tid; k < H4 + EE; k += 256) {
        float v;
        if (k < H2)            v = ct[b * H2 + k];
        else if (k < H2 + HH)  v = hdec[b * HH + (k - H2)];
        else if (k < H4)       v = cdec[b * HH + (k - H2 - HH)];
        else                   v = xbuf[b * EE + (k - H4)];
        p += v * W_pgen[k];
    }
    red[tid] = p; __syncthreads();
    for (int s = 128; s > 0; s >>= 1) { if (tid < s) red[tid] += red[tid + s]; __syncthreads(); }
    if (tid == 0) pgen_out[b] = sigmf(red[0] + b_pgen[0]);
}

// ---------------- K8: pack out1 into bf16 WMMA A-fragment layout -------------
// 16-bit A 16x32 layout: lane L -> m = L%16, half = L/16; elem e -> r = e/2,
// group = r/4, K = kstep*32 + group*16 + half*8 + (r%4)*2 + (e%2)
__global__ void k_packA(const float* __restrict__ out1f, __bf16* __restrict__ wsA) {
    const int idx = blockIdx.x * 256 + threadIdx.x;  // 32768 elements
    const int e = idx & 15;
    const int lane = (idx >> 4) & 31;
    const int ks = (idx >> 9) & 15;
    const int mt = idx >> 13;
    const int m = mt * 16 + (lane & 15);
    const int half = lane >> 4;
    const int r = e >> 1, grp = r >> 2;
    const int K = ks * 32 + grp * 16 + half * 8 + (r & 3) * 2 + (e & 1);
    wsA[idx] = f2bf(out1f[m * HH + K]);
}

// ---------------- K9: WMMA logits GEMM [64,512]x[512,50000] ------------------
__global__ void k_logits(const __bf16* __restrict__ wsA, const float* __restrict__ W2,
                         const float* __restrict__ b2, float* __restrict__ logits) {
    const int lane = threadIdx.x & 31;
    const int wave = threadIdx.x >> 5;
    const int nbase = blockIdx.x * 64 + wave * 16;
    const int ncol = nbase + (lane & 15);
    const int khalf = lane >> 4;
    const bool valid = (ncol < VV);
    v8f acc[4] = {};
    for (int ks = 0; ks < 16; ++ks) {
        v16bf bfrag;
        if (valid) {
            const float* row = W2 + (size_t)ncol * HH;
            const int k0 = ks * 32 + khalf * 8;
#pragma unroll
            for (int j = 0; j < 8; ++j) bfrag[j] = f2bf(row[k0 + j]);
#pragma unroll
            for (int j = 0; j < 8; ++j) bfrag[8 + j] = f2bf(row[k0 + 16 + j]);
        } else {
#pragma unroll
            for (int j = 0; j < 16; ++j) bfrag[j] = f2bf(0.f);
        }
#pragma unroll
        for (int mt = 0; mt < 4; ++mt) {
            const v16bf a = *(const v16bf*)(wsA + (((mt * 16 + ks) * 32 + lane) << 4));
            acc[mt] = __builtin_amdgcn_wmma_f32_16x16x32_bf16(
                false, a, false, bfrag, (short)0, acc[mt], false, false);
        }
    }
    if (valid) {
        const float bias = b2[ncol];
#pragma unroll
        for (int mt = 0; mt < 4; ++mt)
#pragma unroll
            for (int r = 0; r < 8; ++r) {
                const int m = mt * 16 + r + 8 * khalf;
                logits[(size_t)m * VV + ncol] = acc[mt][r] + bias;
            }
    }
}

// ---------------- K10: softmax over V, write p_gen*dist + zero OOV -----------
__global__ void k_softmax_vocab(const float* __restrict__ logits, const float* __restrict__ pgen,
                                float* __restrict__ final_out) {
    __shared__ float red[256];
    const int b = blockIdx.x;
    const int tid = threadIdx.x;
    float m = -3.4e38f;
    for (int v = tid; v < VV; v += 256) m = fmaxf(m, logits[(size_t)b * VV + v]);
    red[tid] = m; __syncthreads();
    for (int s = 128; s > 0; s >>= 1) { if (tid < s) red[tid] = fmaxf(red[tid], red[tid + s]); __syncthreads(); }
    m = red[0]; __syncthreads();
    float s = 0.f;
    for (int v = tid; v < VV; v += 256) s += expf(logits[(size_t)b * VV + v] - m);
    red[tid] = s; __syncthreads();
    for (int q = 128; q > 0; q >>= 1) { if (tid < q) red[tid] += red[tid + q]; __syncthreads(); }
    const float scale = pgen[b] / red[0];
    for (int v = tid; v < VV; v += 256)
        final_out[(size_t)b * VEXT + v] = expf(logits[(size_t)b * VV + v] - m) * scale;
    for (int j = tid; j < NOOV; j += 256) final_out[(size_t)b * VEXT + VV + j] = 0.f;
}

// ---------------- K11: pointer scatter-add -----------------------------------
__global__ void k_scatter(const int* __restrict__ ivoc, const float* __restrict__ attn,
                          const float* __restrict__ pgen, float* __restrict__ final_out) {
    const int g = blockIdx.x * 256 + threadIdx.x;
    if (g >= TKK * BB) return;
    const int t = g >> 6;
    const int b = g & 63;
    const float w = (1.f - pgen[b]) * attn[b * TKK + t];
    const int col = ivoc[t * BB + b];
    atomicAdd(&final_out[(size_t)b * VEXT + col], w);
}

extern "C" void kernel_launch(void* const* d_in, const int* in_sizes, int n_in,
                              void* d_out, int out_size, void* d_ws, size_t ws_size,
                              hipStream_t stream) {
    const int*   ids   = (const int*)d_in[0];
    const float* h0    = (const float*)d_in[1];
    const float* c0    = (const float*)d_in[2];
    const float* EO    = (const float*)d_in[3];
    const float* EF    = (const float*)d_in[4];
    const float* msk   = (const float*)d_in[5];
    const float* ct1   = (const float*)d_in[6];
    const int*   ivoc  = (const int*)d_in[7];
    const float* cov   = (const float*)d_in[8];
    const float* W_emb = (const float*)d_in[10];
    const float* W_ctx = (const float*)d_in[11];
    const float* b_ctx = (const float*)d_in[12];
    const float* W_ih  = (const float*)d_in[13];
    const float* W_hh  = (const float*)d_in[14];
    const float* b_ih  = (const float*)d_in[15];
    const float* b_hh  = (const float*)d_in[16];
    const float* W_at  = (const float*)d_in[17];
    const float* b_at  = (const float*)d_in[18];
    const float* v_at  = (const float*)d_in[19];
    const float* W_pg  = (const float*)d_in[20];
    const float* b_pg  = (const float*)d_in[21];
    const float* W_o1  = (const float*)d_in[22];
    const float* b_o1  = (const float*)d_in[23];
    const float* W_o2  = (const float*)d_in[24];
    const float* b_o2  = (const float*)d_in[25];

    float* out = (float*)d_out;  // flattened tuple, in return order
    float* o_final = out;                      // [64, 50512]
    float* o_h  = o_final + (size_t)BB * VEXT; // [1,64,512]
    float* o_c  = o_h + BB * HH;               // [1,64,512]
    float* o_ct = o_c + BB * HH;               // [64,1024]
    float* o_at = o_ct + BB * H2;              // [64,2048]
    float* o_pg = o_at + BB * TKK;             // [64,1]
    float* o_cv = o_pg + BB;                   // [64,2048]

    float* ws   = (float*)d_ws;
    float* w_x  = ws;                          // 8192
    float* w_df = w_x + BB * EE;               // 65536
    float* w_sc = w_df + BB * H2;              // 131072
    float* w_o1 = w_sc + BB * TKK;             // 32768
    float* w_lg = w_o1 + BB * HH;              // 3,200,000
    __bf16* w_A = (__bf16*)(w_lg + (size_t)BB * VV);  // 32768 bf16 (32B-aligned)

    k_ctx_proj<<<BB, 128, 0, stream>>>(ids, W_emb, ct1, W_ctx, b_ctx, w_x);
    k_lstm<<<128, 256, 0, stream>>>(w_x, h0, c0, W_ih, W_hh, b_ih, b_hh, o_h, o_c);
    k_decfeat<<<256, 256, 0, stream>>>(o_h, o_c, W_at, b_at, w_df);
    k_scores<<<dim3(TKK / TT, BB), 256, 0, stream>>>(EF, w_df, v_at, w_sc);
    k_softmax_attn<<<BB, 256, 0, stream>>>(w_sc, msk, cov, o_at, o_ct, o_cv);
    k_ctxvec<<<dim3(TSPLIT, BB), 256, 0, stream>>>(EO, o_at, o_ct);
    k_pgen_out1<<<BB, 256, 0, stream>>>(o_h, o_c, o_ct, w_x, W_o1, b_o1, W_pg, b_pg, w_o1, o_pg);
    k_packA<<<128, 256, 0, stream>>>(w_o1, w_A);
    k_logits<<<(VV + 63) / 64, 128, 0, stream>>>(w_A, W_o2, b_o2, w_lg);
    k_softmax_vocab<<<BB, 256, 0, stream>>>(w_lg, o_pg, o_final);
    k_scatter<<<(TKK * BB + 255) / 256, 256, 0, stream>>>(ivoc, o_at, o_pg, o_final);
}
